// SparseLinear_14903536517962
// MI455X (gfx1250) — matrix operand order repaired and checked by reading, not compiled, not run on hardware
//
#include <hip/hip_runtime.h>
#include <stdint.h>

#define M_DIM 4096
#define K_DIM 4096
#define B_DIM 8
#define S_DIM 2048

typedef __attribute__((ext_vector_type(16))) __bf16 v16bf;
typedef __attribute__((ext_vector_type(8)))  float  v8f;
typedef __attribute__((ext_vector_type(4)))  unsigned int u32x4;
typedef __attribute__((ext_vector_type(8)))  int i32x8;
typedef __attribute__((ext_vector_type(4)))  int i32x4;

#if defined(__has_builtin)
#if __has_builtin(__builtin_amdgcn_tensor_load_to_lds)
#define HAVE_TDM 1
#endif
#endif
#ifndef HAVE_TDM
#define HAVE_TDM 0
#endif

// ---------- bf16 helpers (RNE) ----------
static __device__ __forceinline__ unsigned short f2bf(float f) {
    union { float f; unsigned u; } c; c.f = f;
    unsigned u = c.u;
    unsigned r = u + 0x7FFFu + ((u >> 16) & 1u);
    return (unsigned short)(r >> 16);
}
static __device__ __forceinline__ float bf2f(unsigned short h) {
    union { unsigned u; float f; } c; c.u = ((unsigned)h) << 16;
    return c.f;
}
// split float4 -> packed hi (4xbf16 in uint2) and lo (residual) pairs
static __device__ __forceinline__ void split4(float4 v, uint2* hp, uint2* lp) {
    unsigned short h0 = f2bf(v.x), h1 = f2bf(v.y), h2 = f2bf(v.z), h3 = f2bf(v.w);
    unsigned short l0 = f2bf(v.x - bf2f(h0));
    unsigned short l1 = f2bf(v.y - bf2f(h1));
    unsigned short l2 = f2bf(v.z - bf2f(h2));
    unsigned short l3 = f2bf(v.w - bf2f(h3));
    hp->x = (unsigned)h0 | ((unsigned)h1 << 16); hp->y = (unsigned)h2 | ((unsigned)h3 << 16);
    lp->x = (unsigned)l0 | ((unsigned)l1 << 16); lp->y = (unsigned)l2 | ((unsigned)l3 << 16);
}

union Frag { v16bf v; uint4 q[2]; };

// ---------- Phase 1: build dense W in workspace ----------
__global__ void zeroW_kernel(float4* __restrict__ w, int n4) {
    int i = blockIdx.x * blockDim.x + threadIdx.x;
    if (i < n4) { float4 z; z.x = z.y = z.z = z.w = 0.f; w[i] = z; }
}

__global__ void scatterW_kernel(const float* __restrict__ vals,
                                const int* __restrict__ rows,
                                const int* __restrict__ cols,
                                float* __restrict__ w, int nnz) {
    int i = blockIdx.x * blockDim.x + threadIdx.x;
    if (i < nnz) {
        atomicAdd(&w[(size_t)rows[i] * K_DIM + cols[i]], vals[i]);
    }
}

// ---------- Phase 1.5: one-time fp32 -> (bf16 hi, bf16 lo) split ----------
__global__ void splitF32_kernel(const float4* __restrict__ src,
                                uint2* __restrict__ hi4,
                                uint2* __restrict__ lo4, int n4) {
    int i = blockIdx.x * blockDim.x + threadIdx.x;
    if (i < n4) {
        uint2 hp, lp;
        split4(src[i], &hp, &lp);
        hi4[i] = hp; lo4[i] = lp;
    }
}

// ---------- TDM tile load: 2D D#, bf16, tile 32(K) x rows, LDS pad -> 80B row stride ----------
#define LSTR 40   // LDS row stride in bf16 elements (64B data + 16B TDM pad)

#if HAVE_TDM
static __device__ __forceinline__ void tdm_load_2d(unsigned lds_byte_addr,
                                                   const unsigned short* gptr,
                                                   unsigned tensor_rows,
                                                   unsigned tile_rows) {
    unsigned long long ga = (unsigned long long)(uintptr_t)gptr;
    u32x4 g0;
    g0.x = 0x1u;                                   // count=1, user mode
    g0.y = lds_byte_addr;                          // lds_addr [63:32]
    g0.z = (unsigned)(ga & 0xFFFFFFFFu);           // global_addr[31:0]
    g0.w = (unsigned)((ga >> 32) & 0x1FFFFFFu) | (2u << 30);  // addr[56:32], type=2
    i32x8 g1;
    // data_size=1 (2B), pad_enable, pad_interval=3 (16 DW = 64B), pad_amount=3 (4 DW = 16B)
    g1[0] = (int)((1u << 16) | (1u << 20) | (3u << 22) | (3u << 25));
    g1[1] = (int)((K_DIM & 0xFFFFu) << 16);                              // tensor_dim0[15:0] @63:48
    g1[2] = (int)(((unsigned)K_DIM >> 16) | ((tensor_rows & 0xFFFFu) << 16)); // dim0 hi, dim1 lo
    g1[3] = (int)((tensor_rows >> 16) | (32u << 16));                    // dim1 hi, tile_dim0=32
    g1[4] = (int)(tile_rows & 0xFFFFu);                                  // tile_dim1 (tile_dim2=0)
    g1[5] = (int)K_DIM;                                                  // tensor_dim0_stride lo32
    g1[6] = 0;                                                           // stride hi / dim1_stride lo
    g1[7] = 0;
    i32x4 z4 = {0, 0, 0, 0};
#if defined(__clang_major__) && (__clang_major__ >= 23)
    i32x8 z8 = {0, 0, 0, 0, 0, 0, 0, 0};
    __builtin_amdgcn_tensor_load_to_lds(g0, g1, z4, z4, z8, 0);
#else
    __builtin_amdgcn_tensor_load_to_lds(g0, g1, z4, z4, 0);
#endif
}
#endif

// ---------- Phase 2 (fast): all-bf16 WMMA GEMM with TDM double-buffered staging ----------
__global__ __launch_bounds__(256)
void spgemm_wmma_tdm_kernel(const unsigned short* __restrict__ xhi,
                            const unsigned short* __restrict__ xlo,
                            const unsigned short* __restrict__ whi,
                            const unsigned short* __restrict__ wlo,
                            const float* __restrict__ bias,
                            float* __restrict__ out) {
    __shared__ unsigned short sWhi[2][64  * LSTR];
    __shared__ unsigned short sWlo[2][64  * LSTR];
    __shared__ unsigned short sXhi[2][128 * LSTR];
    __shared__ unsigned short sXlo[2][128 * LSTR];

    const int t    = threadIdx.x;
    const int b    = blockIdx.z;
    const int m0   = blockIdx.y * 64;
    const int s0   = blockIdx.x * 128;
    const int wave = t >> 5;
    const int lane = t & 31;
    const int hb   = lane >> 4;
    const int l16  = lane & 15;
    const int wm   = wave >> 2;     // 0..1  -> 32 M-rows
    const int wn   = wave & 3;      // 0..3  -> 32 S-cols

    const int xrow0 = b * S_DIM + s0;
    v8f c[2][2] = {};

#if HAVE_TDM
    // prologue: wave 0 DMAs the k0=0 tiles into buffer 0
    if (t < 32) {
        tdm_load_2d((unsigned)(uintptr_t)&sWhi[0][0], whi + (size_t)m0 * K_DIM,    M_DIM,         64);
        tdm_load_2d((unsigned)(uintptr_t)&sWlo[0][0], wlo + (size_t)m0 * K_DIM,    M_DIM,         64);
        tdm_load_2d((unsigned)(uintptr_t)&sXhi[0][0], xhi + (size_t)xrow0 * K_DIM, B_DIM * S_DIM, 128);
        tdm_load_2d((unsigned)(uintptr_t)&sXlo[0][0], xlo + (size_t)xrow0 * K_DIM, B_DIM * S_DIM, 128);
        __builtin_amdgcn_s_wait_tensorcnt(0);
    }
    __syncthreads();
#endif

    int buf = 0;
    for (int k0 = 0; k0 < K_DIM; k0 += 32) {
        const int nk = k0 + 32;
#if HAVE_TDM
        // async-prefetch next tiles into the other buffer while we compute
        if (nk < K_DIM && t < 32) {
            const int nb = buf ^ 1;
            tdm_load_2d((unsigned)(uintptr_t)&sWhi[nb][0], whi + (size_t)m0 * K_DIM + nk,    M_DIM,         64);
            tdm_load_2d((unsigned)(uintptr_t)&sWlo[nb][0], wlo + (size_t)m0 * K_DIM + nk,    M_DIM,         64);
            tdm_load_2d((unsigned)(uintptr_t)&sXhi[nb][0], xhi + (size_t)xrow0 * K_DIM + nk, B_DIM * S_DIM, 128);
            tdm_load_2d((unsigned)(uintptr_t)&sXlo[nb][0], xlo + (size_t)xrow0 * K_DIM + nk, B_DIM * S_DIM, 128);
        }
#else
        // synchronous bf16 staging fallback (no TDM builtin on this toolchain)
        {
            const int nb = buf;
            #pragma unroll
            for (int i = 0; i < 1; ++i) {              // W: 64 rows * 4 uint4/row = 256
                int idx = t; int row = idx >> 2; int colq = idx & 3;
                *(uint4*)(&sWhi[nb][row * LSTR + colq * 8]) =
                    *(const uint4*)(whi + (size_t)(m0 + row) * K_DIM + k0 + colq * 8);
                *(uint4*)(&sWlo[nb][row * LSTR + colq * 8]) =
                    *(const uint4*)(wlo + (size_t)(m0 + row) * K_DIM + k0 + colq * 8);
            }
            #pragma unroll
            for (int i = 0; i < 2; ++i) {              // X: 128 rows * 4 uint4/row = 512
                int idx = t + i * 256; int row = idx >> 2; int colq = idx & 3;
                *(uint4*)(&sXhi[nb][row * LSTR + colq * 8]) =
                    *(const uint4*)(xhi + (size_t)(xrow0 + row) * K_DIM + k0 + colq * 8);
                *(uint4*)(&sXlo[nb][row * LSTR + colq * 8]) =
                    *(const uint4*)(xlo + (size_t)(xrow0 + row) * K_DIM + k0 + colq * 8);
            }
            __syncthreads();
        }
#endif
        // ---- fragments from current buffer (ISA 16-bit A/B layouts) ----
        const unsigned short* pWhi = sWhi[buf];
        const unsigned short* pWlo = sWlo[buf];
        const unsigned short* pXhi = sXhi[buf];
        const unsigned short* pXlo = sXlo[buf];

        Frag ahi[2], alo[2], bhi[2], blo[2];
        #pragma unroll
        for (int mt = 0; mt < 2; ++mt) {
            int base = (wm * 32 + mt * 16 + l16) * LSTR + hb * 8;
            ahi[mt].q[0] = *(const uint4*)(pWhi + base);
            ahi[mt].q[1] = *(const uint4*)(pWhi + base + 16);
            alo[mt].q[0] = *(const uint4*)(pWlo + base);
            alo[mt].q[1] = *(const uint4*)(pWlo + base + 16);
        }
        #pragma unroll
        for (int nt = 0; nt < 2; ++nt) {
            int base = (wn * 32 + nt * 16 + l16) * LSTR + hb * 8;
            bhi[nt].q[0] = *(const uint4*)(pXhi + base);
            bhi[nt].q[1] = *(const uint4*)(pXhi + base + 16);
            blo[nt].q[0] = *(const uint4*)(pXlo + base);
            blo[nt].q[1] = *(const uint4*)(pXlo + base + 16);
        }
        #pragma unroll
        for (int mt = 0; mt < 2; ++mt) {
            #pragma unroll
            for (int nt = 0; nt < 2; ++nt) {
                c[mt][nt] = __builtin_amdgcn_wmma_f32_16x16x32_bf16(
                    false, ahi[mt].v, false, bhi[nt].v, (short)0, c[mt][nt], false, false);
                c[mt][nt] = __builtin_amdgcn_wmma_f32_16x16x32_bf16(
                    false, ahi[mt].v, false, blo[nt].v, (short)0, c[mt][nt], false, false);
                c[mt][nt] = __builtin_amdgcn_wmma_f32_16x16x32_bf16(
                    false, alo[mt].v, false, bhi[nt].v, (short)0, c[mt][nt], false, false);
            }
        }
#if HAVE_TDM
        if (nk < K_DIM && t < 32) __builtin_amdgcn_s_wait_tensorcnt(0);
#endif
        __syncthreads();
        buf ^= 1;
    }

    // ---- epilogue ----
    #pragma unroll
    for (int mt = 0; mt < 2; ++mt) {
        #pragma unroll
        for (int nt = 0; nt < 2; ++nt) {
            int s     = s0 + wn * 32 + nt * 16 + l16;
            int mbase = m0 + wm * 32 + mt * 16 + hb * 8;
            float bs  = bias[s];
            float* op = out + ((size_t)b * M_DIM + mbase) * S_DIM + s;
            #pragma unroll
            for (int r = 0; r < 8; ++r) op[(size_t)r * S_DIM] = c[mt][nt][r] + bs;
        }
    }
}

// ---------- Phase 2 (fallback): fp32 sources, split during LDS staging ----------
__global__ __launch_bounds__(256)
void spgemm_wmma_kernel(const float* __restrict__ x,
                        const float* __restrict__ bias,
                        const float* __restrict__ w,
                        float* __restrict__ out) {
    __shared__ unsigned short sWhi[64  * LSTR];
    __shared__ unsigned short sWlo[64  * LSTR];
    __shared__ unsigned short sXhi[128 * LSTR];
    __shared__ unsigned short sXlo[128 * LSTR];

    const int t    = threadIdx.x;
    const int b    = blockIdx.z;
    const int m0   = blockIdx.y * 64;
    const int s0   = blockIdx.x * 128;
    const int wave = t >> 5;
    const int lane = t & 31;
    const int hb   = lane >> 4;
    const int l16  = lane & 15;
    const int wm   = wave >> 2;
    const int wn   = wave & 3;

    v8f c[2][2] = {};
    const float* wBase = w + (size_t)m0 * K_DIM;
    const float* xBase = x + ((size_t)b * S_DIM + s0) * K_DIM;

    for (int k0 = 0; k0 < K_DIM; k0 += 32) {
        __syncthreads();
        #pragma unroll
        for (int i = 0; i < 2; ++i) {
            int idx = t + i * 256; int row = idx >> 3; int colv = idx & 7;
            float4 v = *(const float4*)(wBase + (size_t)row * K_DIM + k0 + colv * 4);
            uint2 hp, lp; split4(v, &hp, &lp);
            *(uint2*)(&sWhi[row * LSTR + colv * 4]) = hp;
            *(uint2*)(&sWlo[row * LSTR + colv * 4]) = lp;
        }
        #pragma unroll
        for (int i = 0; i < 4; ++i) {
            int idx = t + i * 256; int row = idx >> 3; int colv = idx & 7;
            float4 v = *(const float4*)(xBase + (size_t)row * K_DIM + k0 + colv * 4);
            uint2 hp, lp; split4(v, &hp, &lp);
            *(uint2*)(&sXhi[row * LSTR + colv * 4]) = hp;
            *(uint2*)(&sXlo[row * LSTR + colv * 4]) = lp;
        }
        __syncthreads();

        Frag ahi[2], alo[2], bhi[2], blo[2];
        #pragma unroll
        for (int mt = 0; mt < 2; ++mt) {
            int base = (wm * 32 + mt * 16 + l16) * LSTR + hb * 8;
            ahi[mt].q[0] = *(const uint4*)(sWhi + base);
            ahi[mt].q[1] = *(const uint4*)(sWhi + base + 16);
            alo[mt].q[0] = *(const uint4*)(sWlo + base);
            alo[mt].q[1] = *(const uint4*)(sWlo + base + 16);
        }
        #pragma unroll
        for (int nt = 0; nt < 2; ++nt) {
            int base = (wn * 32 + nt * 16 + l16) * LSTR + hb * 8;
            bhi[nt].q[0] = *(const uint4*)(sXhi + base);
            bhi[nt].q[1] = *(const uint4*)(sXhi + base + 16);
            blo[nt].q[0] = *(const uint4*)(sXlo + base);
            blo[nt].q[1] = *(const uint4*)(sXlo + base + 16);
        }
        #pragma unroll
        for (int mt = 0; mt < 2; ++mt) {
            #pragma unroll
            for (int nt = 0; nt < 2; ++nt) {
                c[mt][nt] = __builtin_amdgcn_wmma_f32_16x16x32_bf16(
                    false, ahi[mt].v, false, bhi[nt].v, (short)0, c[mt][nt], false, false);
                c[mt][nt] = __builtin_amdgcn_wmma_f32_16x16x32_bf16(
                    false, ahi[mt].v, false, blo[nt].v, (short)0, c[mt][nt], false, false);
                c[mt][nt] = __builtin_amdgcn_wmma_f32_16x16x32_bf16(
                    false, alo[mt].v, false, bhi[nt].v, (short)0, c[mt][nt], false, false);
            }
        }
    }

    #pragma unroll
    for (int mt = 0; mt < 2; ++mt) {
        #pragma unroll
        for (int nt = 0; nt < 2; ++nt) {
            int s     = s0 + wn * 32 + nt * 16 + l16;
            int mbase = m0 + wm * 32 + mt * 16 + hb * 8;
            float bs  = bias[s];
            float* op = out + ((size_t)b * M_DIM + mbase) * S_DIM + s;
            #pragma unroll
            for (int r = 0; r < 8; ++r) op[(size_t)r * S_DIM] = c[mt][nt][r] + bs;
        }
    }
}

extern "C" void kernel_launch(void* const* d_in, const int* in_sizes, int n_in,
                              void* d_out, int out_size, void* d_ws, size_t ws_size,
                              hipStream_t stream) {
    (void)n_in; (void)out_size;
    const float* x      = (const float*)d_in[0];
    const float* values = (const float*)d_in[1];
    const float* bias   = (const float*)d_in[2];
    const int*   rows   = (const int*)d_in[3];
    const int*   cols   = (const int*)d_in[4];
    float*       out    = (float*)d_out;
    float*       w      = (float*)d_ws;                 // dense W fp32: 64 MiB
    const int    nnz    = in_sizes[1];

    const size_t MB = 1024 * 1024;
    const int nW4 = (M_DIM * K_DIM) / 4;                // 4Mi float4
    const int nX4 = (B_DIM * S_DIM * K_DIM) / 4;        // 16Mi float4

    // Phase 1: W = scatter_add(COO)
    zeroW_kernel<<<(nW4 + 255) / 256, 256, 0, stream>>>((float4*)w, nW4);
    scatterW_kernel<<<(nnz + 255) / 256, 256, 0, stream>>>(values, rows, cols, w, nnz);

    dim3 grid(S_DIM / 128, M_DIM / 64, B_DIM);

    if (ws_size >= 384 * MB) {
        // Fast path: one-time bf16 hi/lo split of W and x, then TDM-staged WMMA GEMM.
        unsigned short* whi = (unsigned short*)((char*)d_ws + 64 * MB);
        unsigned short* wlo = (unsigned short*)((char*)d_ws + 96 * MB);
        unsigned short* xhi = (unsigned short*)((char*)d_ws + 128 * MB);
        unsigned short* xlo = (unsigned short*)((char*)d_ws + 256 * MB);
        splitF32_kernel<<<(nW4 + 255) / 256, 256, 0, stream>>>(
            (const float4*)w, (uint2*)whi, (uint2*)wlo, nW4);
        splitF32_kernel<<<(nX4 + 255) / 256, 256, 0, stream>>>(
            (const float4*)x, (uint2*)xhi, (uint2*)xlo, nX4);
        spgemm_wmma_tdm_kernel<<<grid, 256, 0, stream>>>(xhi, xlo, whi, wlo, bias, out);
    } else {
        // Fallback: split during LDS staging (64 MiB workspace only).
        spgemm_wmma_kernel<<<grid, 256, 0, stream>>>(x, bias, w, out);
    }
}
